// MambaSemanticMapper_Dual_50208167690332
// MI455X (gfx1250) — compile-verified, matrix operand order checked by reading, not run on hardware
//
#include <hip/hip_runtime.h>

// ---------------------------------------------------------------------------
// Bidirectional Mamba block for gfx1250 (MI455X), wave32 + WMMA.
//   B=4, L=2048, CLIP=512, D_MODEL=256, D_INNER=512, D_STATE=16, DT_RANK=16
// GEMM: templated, 8 waves/block, block tile 64(M) x 128(N); each wave owns a
// 16x64 strip = 4 accumulators sharing one B fragment. A staged in LDS with
// 20-dword (80 B) row stride so every fragment gather is 2x ds_load_b128;
// all 4 A fragments are gathered into disjoint register blocks BEFORE the 4
// back-to-back WMMAs (no WMMA->VALU WAR hazard NOPs).
// ---------------------------------------------------------------------------

typedef __attribute__((ext_vector_type(16))) _Float16 v16h;
typedef __attribute__((ext_vector_type(8)))  float    v8f;
typedef unsigned int u32;

#define NB   4
#define LSEQ 2048
#define NTOK (NB * LSEQ)          // 8192
#define CLIP 512
#define DMODEL 256
#define DINNER 512
#define DSTATE 16
#define DTRANK 16
#define XDBL_STRIDE 128           // 48 real cols padded to 128 (one N-block)

struct __align__(16) U4 { u32 x, y, z, w; };

// --------------------------- weight fragment packing ------------------------
// Pack W (f32, row-major [Nreal, K], zero-padded to Npad rows) into WMMA
// B-fragment layout: dword index = ((tileN*(K/32)+kc)*32 + lane)*8 + v
// lane 0..15 -> col n = tileN*16 + lane, K-half group g=0; lane 16..31 -> g=1.
// VGPR v<4 : k = kc*32 + g*8 + 2v ; v>=4 : k = kc*32 + 16 + g*8 + 2(v-4).
__launch_bounds__(256)
__global__ void pack_b_kernel(const float* __restrict__ W, u32* __restrict__ Bp,
                              int Nreal, int Npad, int K) {
  int idx = blockIdx.x * 256 + threadIdx.x;
  int total = (Npad >> 4) * (K >> 5) * 256;
  if (idx >= total) return;
  int v    = idx & 7;
  int lane = (idx >> 3) & 31;
  int blk  = idx >> 8;
  int kch  = K >> 5;
  int kc   = blk % kch;
  int tn   = blk / kch;
  int n    = tn * 16 + (lane & 15);
  int g    = lane >> 4;
  int c    = (v < 4) ? (g * 8 + 2 * v) : (16 + g * 8 + 2 * (v - 4));
  int k    = kc * 32 + c;
  float f0 = 0.f, f1 = 0.f;
  if (n < Nreal) { f0 = W[(size_t)n * K + k]; f1 = W[(size_t)n * K + k + 1]; }
  union { _Float16 h[2]; u32 u; } p;
  p.h[0] = (_Float16)f0; p.h[1] = (_Float16)f1;
  Bp[idx] = p.u;
}

// ------------------------------- WMMA GEMM ----------------------------------
// C[M,N] (f32 acc) = A[M,K] * Bpacked^T (+bias).
// Block = 256 threads = 8 waves; wave w owns N-tile w across 4 M-sub-tiles.
template <int AISF32, int OUTF16, int ACCUM>
__launch_bounds__(256)
__global__ void gemm_wmma_kernel(const void* __restrict__ Aptr,
                                 const u32* __restrict__ Bp,
                                 const float* __restrict__ bias,
                                 void* __restrict__ Cptr,
                                 int M, int N, int K) {
  // 64 rows x 32 halfs; each row = 5 U4 slots (4 data + 1 pad) = 20 dwords.
  // 16B-aligned rows; bank map 20*r mod 64 is conflict-free for r=0..15.
  __shared__ U4 smemA[64 * 5];
  const int tid   = threadIdx.x;
  const int wave  = tid >> 5;
  const int lane  = tid & 31;
  const int mBase = blockIdx.x * 64;
  const int nBase = blockIdx.y * 128 + wave * 16;
  const int kch   = K >> 5;
  const int sRow  = tid >> 2;                 // staging: row 0..63
  const int sSlot = tid & 3;                  // staging: U4 slot 0..3
  const int g     = lane >> 4;
  const int r     = lane & 15;
  const float*    Af = (const float*)Aptr;
  const _Float16* Ah = (const _Float16*)Aptr;

  v8f acc[4];
#pragma unroll
  for (int ms = 0; ms < 4; ++ms) acc[ms] = (v8f){0.f,0.f,0.f,0.f,0.f,0.f,0.f,0.f};

  for (int kc = 0; kc < kch; ++kc) {
    const int kb = kc * 32;
    // ---- stage A tile 64x32 halfs (each thread: one 16B slot = 8 halfs) ----
    U4 d;
    if (AISF32) {
      const float* src = Af + (size_t)(mBase + sRow) * K + kb + sSlot * 8;
      union { _Float16 h[2]; u32 u; } p0, p1, p2, p3;
      p0.h[0] = (_Float16)src[0]; p0.h[1] = (_Float16)src[1];
      p1.h[0] = (_Float16)src[2]; p1.h[1] = (_Float16)src[3];
      p2.h[0] = (_Float16)src[4]; p2.h[1] = (_Float16)src[5];
      p3.h[0] = (_Float16)src[6]; p3.h[1] = (_Float16)src[7];
      d.x = p0.u; d.y = p1.u; d.z = p2.u; d.w = p3.u;
    } else {
      d = *(const U4*)(Ah + (size_t)(mBase + sRow) * K + kb + sSlot * 8);
    }
    smemA[sRow * 5 + sSlot] = d;              // ds_store_b128
    __syncthreads();

    // ---- B fragment: 8 contiguous dwords per lane (pre-packed) ----
    union { v16h v; U4 q[2]; } bfrag;
    const u32* bsrc = Bp + ((size_t)((nBase >> 4) * kch + kc) * 32 + lane) * 8;
    __builtin_prefetch(bsrc + 256, 0, 1);     // next K-chunk of this N-tile
    bfrag.q[0] = *(const U4*)bsrc;
    bfrag.q[1] = *(const U4*)(bsrc + 4);

    // ---- gather ALL 4 A fragments first (disjoint regs -> no WMMA hazards) ----
    union { v16h v; U4 q[2]; } afrag[4];
#pragma unroll
    for (int ms = 0; ms < 4; ++ms) {
      const int rowb = ms * 16 + r;
      afrag[ms].q[0] = smemA[rowb * 5 + g];       // dwords c/2 = g*4 .. g*4+3
      afrag[ms].q[1] = smemA[rowb * 5 + 2 + g];   // dwords c/2 = 8+g*4 .. +3
    }
    // ---- 4 back-to-back WMMAs reusing one B fragment ----
#pragma unroll
    for (int ms = 0; ms < 4; ++ms) {
      acc[ms] = __builtin_amdgcn_wmma_f32_16x16x32_f16(
          /*neg_a=*/false, afrag[ms].v, /*neg_b=*/false, bfrag.v,
          /*c_mod=*/(short)0, acc[ms], /*reuse_a=*/false, /*reuse_b=*/false);
    }
    __syncthreads();
  }

  // ---- store C: lane r = col, VGPR v -> row v + 8*g, per M-sub-tile ----
  const int n = nBase + r;
  const float bv = bias ? bias[n] : 0.f;
#pragma unroll
  for (int ms = 0; ms < 4; ++ms) {
    if (OUTF16) {
      _Float16* C16 = (_Float16*)Cptr;
#pragma unroll
      for (int vv = 0; vv < 8; ++vv) {
        int m = mBase + ms * 16 + vv + 8 * g;
        C16[(size_t)m * N + n] = (_Float16)(acc[ms][vv] + bv);
      }
    } else {
      float* C = (float*)Cptr;
#pragma unroll
      for (int vv = 0; vv < 8; ++vv) {
        int m = mBase + ms * 16 + vv + 8 * g;
        size_t off = (size_t)m * N + n;
        float val = acc[ms][vv] + bv;
        if (ACCUM) val += C[off];
        C[off] = val;
      }
    }
  }
}

// ----------------------- depthwise causal conv + SiLU -----------------------
// fwd: xc[t,d] = silu(b[d] + sum_j w[j,d]*xi[t-3+j,d])
// rev: xc[t,d] = silu(b[d] + sum_j w[j,d]*xi[t+3-j,d])   (flip-equivalent)
__launch_bounds__(256)
__global__ void conv_silu_kernel(const _Float16* __restrict__ xz,  // [NTOK, 2*DINNER]
                                 const float* __restrict__ cw,     // [4, DINNER]
                                 const float* __restrict__ cb,     // [DINNER]
                                 _Float16* __restrict__ xc,        // [NTOK, DINNER]
                                 int rev) {
  int idx = blockIdx.x * 256 + threadIdx.x;   // NTOK*DINNER
  int d  = idx & (DINNER - 1);
  int t  = idx >> 9;
  int b  = t >> 11;
  int tl = t & (LSEQ - 1);
  float a = cb[d];
#pragma unroll
  for (int j = 0; j < 4; ++j) {
    int tt = rev ? (tl + 3 - j) : (tl - 3 + j);
    if (tt >= 0 && tt < LSEQ) {
      float xv = (float)xz[(size_t)((b << 11) + tt) * (2 * DINNER) + d];
      a += cw[j * DINNER + d] * xv;
    }
  }
  float s = a / (1.f + __expf(-a));
  xc[idx] = (_Float16)s;
}

// --------------------------- fused selective scan ---------------------------
// One lane per (batch, channel): 16 f32 states in registers; per step fuses the
// dt projection (16 MACs) + softplus + 16x exp-discretization + state update.
__launch_bounds__(256)
__global__ void scan_kernel(const float* __restrict__ xdbl,   // [NTOK, XDBL_STRIDE]
                            const _Float16* __restrict__ xc,  // [NTOK, DINNER]
                            const float* __restrict__ dt_w,   // [DINNER, DTRANK]
                            const float* __restrict__ dt_b,   // [DINNER]
                            const float* __restrict__ A_log,  // [DINNER, DSTATE]
                            float* __restrict__ y,            // [NTOK, DINNER]
                            int rev) {
  int idx = blockIdx.x * 256 + threadIdx.x;   // NB*DINNER = 2048
  int d = idx & (DINNER - 1);
  int b = idx >> 9;
  float w[DTRANK], nA[DSTATE], h[DSTATE];
#pragma unroll
  for (int s = 0; s < DSTATE; ++s) {
    w[s]  = dt_w[d * DTRANK + s];
    nA[s] = -__expf(A_log[d * DSTATE + s]);
    h[s]  = 0.f;
  }
  const float db = dt_b[d];
  for (int step = 0; step < LSEQ; ++step) {
    int tl = rev ? (LSEQ - 1 - step) : step;
    size_t t = (size_t)(b * LSEQ + tl);
    const float* row = xdbl + t * XDBL_STRIDE;   // [0..15]=dt, [16..31]=B, [32..47]=C
    float dtp = db;
#pragma unroll
    for (int rr = 0; rr < DTRANK; ++rr) dtp += row[rr] * w[rr];
    float dt = (dtp > 20.f) ? dtp : __logf(1.f + __expf(dtp));   // softplus
    float u  = (float)xc[t * DINNER + d];
    float du = dt * u;
    float yv = 0.f;
#pragma unroll
    for (int s = 0; s < DSTATE; ++s) {
      float a = __expf(dt * nA[s]);
      h[s] = a * h[s] + du * row[16 + s];
      yv  += h[s] * row[32 + s];
    }
    y[t * DINNER + d] = yv;
  }
}

// ------------------------------ gating --------------------------------------
__launch_bounds__(256)
__global__ void gate_kernel(const float* __restrict__ y,
                            const _Float16* __restrict__ xc,
                            const _Float16* __restrict__ xz,   // z = cols [512..1024)
                            const float* __restrict__ Dp,
                            _Float16* __restrict__ gA) {
  int idx = blockIdx.x * 256 + threadIdx.x;   // NTOK*DINNER
  int d = idx & (DINNER - 1);
  size_t t = (size_t)(idx >> 9);
  float u  = (float)xc[idx];
  float z  = (float)xz[t * (2 * DINNER) + DINNER + d];
  float yv = y[idx] + Dp[d] * u;
  float sz = z / (1.f + __expf(-z));
  gA[idx] = (_Float16)(yv * sz);
}

// ------------------------------ mean pool -----------------------------------
__launch_bounds__(256)
__global__ void pool_kernel(const float* __restrict__ outSeq, float* __restrict__ pooled) {
  int idx = blockIdx.x * 256 + threadIdx.x;   // NB*DMODEL = 1024
  int c = idx & (DMODEL - 1);
  int b = idx >> 8;
  float s = 0.f;
  for (int t = 0; t < LSEQ; ++t) s += outSeq[((size_t)b * LSEQ + t) * DMODEL + c];
  pooled[idx] = s * (1.f / (float)LSEQ);
}

// ------------------------------ launcher ------------------------------------
extern "C" void kernel_launch(void* const* d_in, const int* in_sizes, int n_in,
                              void* d_out, int out_size, void* d_ws, size_t ws_size,
                              hipStream_t stream) {
  (void)n_in; (void)out_size; (void)ws_size;
  const float* text = (const float*)d_in[0];
  const float* w_in = (const float*)d_in[1];
  const float* b_in = (const float*)d_in[2];

  // Param dict flattening order: insertion (in_proj first, 262144 elems) vs
  // alphabetical pytree (A_log first, 8192 elems). Detect via in_sizes[3].
  const bool alpha = (in_sizes[3] == DINNER * DSTATE);
  const int o_in_proj  = alpha ? 6 : 0;
  const int o_conv_w   = alpha ? 3 : 1;
  const int o_conv_b   = alpha ? 2 : 2;
  const int o_x_proj   = alpha ? 8 : 3;
  const int o_dt_w     = alpha ? 5 : 4;
  const int o_dt_b     = alpha ? 4 : 5;
  const int o_A_log    = alpha ? 0 : 6;
  const int o_D        = alpha ? 1 : 7;
  const int o_out_proj = alpha ? 7 : 8;

  // ---- workspace carve-up (reused across the two directions) ----
  char* ws = (char*)d_ws;
  auto carve = [&](size_t bytes) {
    char* p = ws; ws += (bytes + 255) & ~(size_t)255; return p;
  };
  u32*      wp_in   = (u32*)     carve((size_t)(DMODEL/16)*(CLIP/32)*256*4);     // 256 KB
  u32*      bp_inp  = (u32*)     carve((size_t)(2*DINNER/16)*(DMODEL/32)*256*4); // 512 KB
  u32*      bp_xp   = (u32*)     carve((size_t)(XDBL_STRIDE/16)*(DINNER/32)*256*4);
  u32*      bp_outp = (u32*)     carve((size_t)(DMODEL/16)*(DINNER/32)*256*4);
  _Float16* x16     = (_Float16*)carve((size_t)NTOK*DMODEL*2);                   // 4 MB
  _Float16* xz16    = (_Float16*)carve((size_t)NTOK*2*DINNER*2);                 // 16 MB
  _Float16* xc16    = (_Float16*)carve((size_t)NTOK*DINNER*2);                   // 8 MB
  float*    xdbl    = (float*)   carve((size_t)NTOK*XDBL_STRIDE*4);              // 4 MB
  float*    ybuf    = (float*)   carve((size_t)NTOK*DINNER*4);                   // 16 MB
  _Float16* gA16    = (_Float16*)carve((size_t)NTOK*DINNER*2);                   // 8 MB

  float* out_seq    = (float*)d_out;
  float* out_pooled = out_seq + (size_t)NTOK * DMODEL;

  auto packN = [](int Npad, int K) { return ((Npad >> 4) * (K >> 5) * 256 + 255) / 256; };

  // ---- proj_in: x16 = f16(text @ w_in^T + b_in) ----
  pack_b_kernel<<<packN(DMODEL, CLIP), 256, 0, stream>>>(w_in, wp_in, DMODEL, DMODEL, CLIP);
  gemm_wmma_kernel<1, 1, 0><<<dim3(NTOK/64, DMODEL/128), 256, 0, stream>>>(
      text, wp_in, b_in, x16, NTOK, DMODEL, CLIP);

  for (int dir = 0; dir < 2; ++dir) {
    const int base = 3 + dir * 9;
    const float* in_proj  = (const float*)d_in[base + o_in_proj];
    const float* conv_w   = (const float*)d_in[base + o_conv_w];
    const float* conv_b   = (const float*)d_in[base + o_conv_b];
    const float* x_proj   = (const float*)d_in[base + o_x_proj];
    const float* dt_w     = (const float*)d_in[base + o_dt_w];
    const float* dt_b     = (const float*)d_in[base + o_dt_b];
    const float* A_log    = (const float*)d_in[base + o_A_log];
    const float* Dp       = (const float*)d_in[base + o_D];
    const float* out_proj = (const float*)d_in[base + o_out_proj];

    pack_b_kernel<<<packN(2*DINNER, DMODEL), 256, 0, stream>>>(
        in_proj, bp_inp, 2*DINNER, 2*DINNER, DMODEL);
    pack_b_kernel<<<packN(XDBL_STRIDE, DINNER), 256, 0, stream>>>(
        x_proj, bp_xp, DTRANK + 2*DSTATE, XDBL_STRIDE, DINNER);
    pack_b_kernel<<<packN(DMODEL, DINNER), 256, 0, stream>>>(
        out_proj, bp_outp, DMODEL, DMODEL, DINNER);

    // xz = x @ in_proj^T
    gemm_wmma_kernel<0, 1, 0><<<dim3(NTOK/64, (2*DINNER)/128), 256, 0, stream>>>(
        x16, bp_inp, nullptr, xz16, NTOK, 2*DINNER, DMODEL);
    // depthwise conv + silu (direction-aware, no physical flip)
    conv_silu_kernel<<<(NTOK*DINNER)/256, 256, 0, stream>>>(xz16, conv_w, conv_b, xc16, dir);
    // xdbl = xc @ x_proj^T  (N padded 48 -> 128)
    gemm_wmma_kernel<0, 0, 0><<<dim3(NTOK/64, XDBL_STRIDE/128), 256, 0, stream>>>(
        xc16, bp_xp, nullptr, xdbl, NTOK, XDBL_STRIDE, DINNER);
    // fused dt-proj + softplus + selective scan
    scan_kernel<<<(NB*DINNER)/256, 256, 0, stream>>>(xdbl, xc16, dt_w, dt_b, A_log, ybuf, dir);
    // gate: (y + D*u) * silu(z)
    gate_kernel<<<(NTOK*DINNER)/256, 256, 0, stream>>>(ybuf, xc16, xz16, Dp, gA16);
    // out += gA @ out_proj^T  (dir 0 writes, dir 1 accumulates)
    if (dir == 0) {
      gemm_wmma_kernel<0, 0, 0><<<dim3(NTOK/64, DMODEL/128), 256, 0, stream>>>(
          gA16, bp_outp, nullptr, out_seq, NTOK, DMODEL, DINNER);
    } else {
      gemm_wmma_kernel<0, 0, 1><<<dim3(NTOK/64, DMODEL/128), 256, 0, stream>>>(
          gA16, bp_outp, nullptr, out_seq, NTOK, DMODEL, DINNER);
    }
  }

  pool_kernel<<<(NB*DMODEL)/256, 256, 0, stream>>>(out_seq, out_pooled);
}
